// LSTM_22677427323268
// MI455X (gfx1250) — matrix operand order repaired and checked by reading, not compiled, not run on hardware
//
#include <hip/hip_runtime.h>
#include <hip/hip_bf16.h>
#include <math.h>

// ---------------------------------------------------------------------------
// Problem constants (match reference)
// ---------------------------------------------------------------------------
#define B_DIM  16384
#define I_DIM  1024
#define H_DIM  1024
#define O_DIM  1024
#define K_SUM  2048   // I + H
#define FOUR_H 4096

// GEMM tiling
#define BM1 128       // rows per block, kernel 1
#define BN1 32        // h-columns per block, kernel 1 (x4 gates internally)
#define BM2 16        // rows per block, kernel 2 (full 1024 cols per block)
#define KT  32        // K step == WMMA K for bf16
#define LDK 40        // padded LDS k-stride in bf16 (80B: 16B-aligned, conflict-skewed)

typedef __attribute__((ext_vector_type(16))) __bf16 v16bf;
typedef __attribute__((ext_vector_type(8)))  float  v8f;
typedef int v4i __attribute__((vector_size(16)));   // builtin's pointee type

// ---------------------------------------------------------------------------
// gfx1250 async global->LDS copy (ASYNCcnt-tracked), with safe fallbacks
// ---------------------------------------------------------------------------
#if __has_builtin(__builtin_amdgcn_global_load_async_to_lds_b128)
  #define HAVE_ASYNC 1
#endif

#ifdef HAVE_ASYNC
  #if __has_builtin(__builtin_amdgcn_s_wait_asynccnt)
    #define WAIT_ASYNC(n) __builtin_amdgcn_s_wait_asynccnt(n)
  #else
    #define WAIT_ASYNC(n) asm volatile("s_wait_asynccnt " #n ::: "memory")
  #endif
#else
  #define WAIT_ASYNC(n) ((void)0)
#endif

__device__ __forceinline__ void copy16(const __bf16* g, __bf16* l) {
#ifdef HAVE_ASYNC
  __builtin_amdgcn_global_load_async_to_lds_b128((v4i*)g, (v4i*)l, 0, 0);
#else
  *reinterpret_cast<uint4*>(l) = *reinterpret_cast<const uint4*>(g);
#endif
}

// f32 -> bf16 round-to-nearest-even
__device__ __forceinline__ __bf16 f2bf(float f) {
  unsigned u = __builtin_bit_cast(unsigned, f);
  u += 0x7FFFu + ((u >> 16) & 1u);
  unsigned short hs = (unsigned short)(u >> 16);
  return __builtin_bit_cast(__bf16, hs);
}
__device__ __forceinline__ unsigned pk2(float a, float b) {
  return (unsigned)__builtin_bit_cast(unsigned short, f2bf(a)) |
         ((unsigned)__builtin_bit_cast(unsigned short, f2bf(b)) << 16);
}
__device__ __forceinline__ float sigmoidf_(float x) {
  return 1.0f / (1.0f + __expf(-x));
}

// ---------------------------------------------------------------------------
// Pre-pass 1: Acat = bf16(cat(x,h))  [B][2048]  (convert ONCE, not per tile)
// ---------------------------------------------------------------------------
__global__ __launch_bounds__(256) void cvt_cat_kernel(
    const float* __restrict__ x, const float* __restrict__ h,
    __bf16* __restrict__ dst)
{
  const size_t i = ((size_t)blockIdx.x * 256 + threadIdx.x) * 8; // 8 elems/thread
  const int k   = (int)(i & (K_SUM - 1));
  const int row = (int)(i >> 11);
  const float* src = (k < I_DIM) ? &x[(size_t)row * I_DIM + k]
                                 : &h[(size_t)row * H_DIM + (k - I_DIM)];
  const float4 a = reinterpret_cast<const float4*>(src)[0];
  const float4 b = reinterpret_cast<const float4*>(src)[1];
  uint4 o;
  o.x = pk2(a.x, a.y); o.y = pk2(a.z, a.w);
  o.z = pk2(b.x, b.y); o.w = pk2(b.z, b.w);
  *reinterpret_cast<uint4*>(&dst[i]) = o;
}

// ---------------------------------------------------------------------------
// Pre-pass 2: transpose+convert  src f32 [K][N]  ->  dst bf16 [N][K]
// (makes weight tiles contiguous along K: perfect for b128 async copies AND
//  for the wave32 B-fragment layout, which wants [n][k] dword pairs)
// ---------------------------------------------------------------------------
__global__ __launch_bounds__(256) void tconv_kernel(
    const float* __restrict__ src, __bf16* __restrict__ dst, int K, int N)
{
  __shared__ float t[32][33];
  const int tid = threadIdx.x;
  const int n0 = blockIdx.x * 32, k0 = blockIdx.y * 32;
  for (int i = tid; i < 1024; i += 256) {
    const int r = i >> 5, c = i & 31;
    t[r][c] = src[(size_t)(k0 + r) * N + n0 + c];
  }
  __syncthreads();
  for (int i = tid; i < 1024; i += 256) {
    const int n = i >> 5, k = i & 31;
    dst[(size_t)(n0 + n) * K + k0 + k] = f2bf(t[k][n]);
  }
}

// ---------------------------------------------------------------------------
// Kernel 1: gates = Acat @ Wg + bg, fused LSTM pointwise.
// bf16 operands streamed via async global->LDS b128 copies, double-buffered.
// Each block: 128 rows x 32 h-cols x 4 gates; 8 waves x (4 gates x 2 tiles).
// ---------------------------------------------------------------------------
__global__ __launch_bounds__(256) void lstm_gates_kernel(
    const __bf16* __restrict__ Acat,   // [B][2048]
    const __bf16* __restrict__ WgT,    // [4096][2048]  (col-major weights)
    const float*  __restrict__ c, const float* __restrict__ bg,
    float* __restrict__ h_new, float* __restrict__ c_new,
    __bf16* __restrict__ h_new_bf)     // [B][1024] for kernel 2
{
  __shared__ __attribute__((aligned(16))) __bf16 Al[2][BM1][LDK]; // 20 KB
  __shared__ __attribute__((aligned(16))) __bf16 Bl[2][128][LDK]; // 20 KB (4g x 32c)

  const int tid  = threadIdx.x;
  const int lane = tid & 31;
  const int wave = tid >> 5;
  const int gsel = lane >> 4;
  const int nl   = lane & 15;
  const int rowBase = blockIdx.x * BM1;
  const int colBase = blockIdx.y * BN1;

  v8f acc[4][2];
  #pragma unroll
  for (int g = 0; g < 4; ++g)
    #pragma unroll
    for (int t = 0; t < 2; ++t)
      acc[g][t] = (v8f){0.f,0.f,0.f,0.f,0.f,0.f,0.f,0.f};

  // stage one K-tile (k0) into buffer `b`: 512 16B chunks each for A and B
  auto stage = [&](int b, int k0) {
    #pragma unroll
    for (int j = 0; j < 2; ++j) {
      const int chunk = tid + 256 * j;
      const int row = chunk >> 2, ck = (chunk & 3) * 8;
      // A: rows of Acat
      copy16(&Acat[(size_t)(rowBase + row) * K_SUM + k0 + ck], &Al[b][row][ck]);
      // B: WgT rows are gate columns; LDS row = gate*32 + col
      const int gate = row >> 5, col = row & 31;
      copy16(&WgT[(size_t)(gate * H_DIM + colBase + col) * K_SUM + k0 + ck],
             &Bl[b][row][ck]);
    }
  };

  stage(0, 0);
  const int NK = K_SUM / KT;               // 64
  for (int ks = 0; ks < NK; ++ks) {
    const int cur = ks & 1;
    if (ks + 1 < NK) {
      stage(1 - cur, (ks + 1) * KT);       // 4 async copies in flight
      WAIT_ASYNC(4);                       // retire the 4 for buffer `cur`
    } else {
      WAIT_ASYNC(0);
    }
    __syncthreads();

    // A fragment: 16x32 bf16 (contiguous k = 16*(lane/16) + e, same order as B)
    v16bf afrag;
    #pragma unroll
    for (int e = 0; e < 16; ++e) {
      const int k = 16 * gsel + e;
      afrag[e] = Al[cur][wave * 16 + nl][k];
    }
    #pragma unroll
    for (int gate = 0; gate < 4; ++gate) {
      #pragma unroll
      for (int t = 0; t < 2; ++t) {
        v16bf bfrag;
        #pragma unroll
        for (int e = 0; e < 16; ++e)
          bfrag[e] = Bl[cur][gate * 32 + t * 16 + nl][16 * gsel + e];
        acc[gate][t] = __builtin_amdgcn_wmma_f32_16x16x32_bf16(
            false, afrag, false, bfrag, (short)0, acc[gate][t], false, false);
      }
    }
    __syncthreads();
  }

  // epilogue: bias + activations + cell update, straight to HBM
  #pragma unroll
  for (int t = 0; t < 2; ++t) {
    const int n  = colBase + t * 16 + nl;
    const float bf_ = bg[0 * H_DIM + n];
    const float bi_ = bg[1 * H_DIM + n];
    const float bu_ = bg[2 * H_DIM + n];
    const float bo_ = bg[3 * H_DIM + n];
    #pragma unroll
    for (int r = 0; r < 8; ++r) {
      const int m = rowBase + wave * 16 + gsel * 8 + r;
      const float fg = sigmoidf_(acc[0][t][r] + bf_);
      const float ig = sigmoidf_(acc[1][t][r] + bi_);
      const float ug = tanhf    (acc[2][t][r] + bu_);
      const float og = sigmoidf_(acc[3][t][r] + bo_);
      const float cn = fg * c[(size_t)m * H_DIM + n] + ig * ug;
      const float hn = og * tanhf(cn);
      c_new[(size_t)m * H_DIM + n]    = cn;
      h_new[(size_t)m * H_DIM + n]    = hn;
      h_new_bf[(size_t)m * H_DIM + n] = f2bf(hn);
    }
  }
}

// ---------------------------------------------------------------------------
// Kernel 2: logits = h_new @ Wout + bout, fused log_softmax.
// 16 rows x full 1024 cols per block; logits reduction stays in LDS.
// ---------------------------------------------------------------------------
union Smem2 {
  __bf16 b[O_DIM][LDK];          // bf16 WoutT staging (80 KB)
  float  logits[BM2][O_DIM];     // reused after K-loop (64 KB)
};

__global__ __launch_bounds__(256) void out_proj_kernel(
    const __bf16* __restrict__ h_new_bf,  // [B][1024]
    const __bf16* __restrict__ WoutT,     // [1024][1024] (col-major weights)
    const float* __restrict__ bout, float* __restrict__ out)
{
  __shared__ __attribute__((aligned(16))) __bf16 Asm[BM2][LDK];
  __shared__ __attribute__((aligned(16))) Smem2 sm;

  const int tid  = threadIdx.x;
  const int lane = tid & 31;
  const int wave = tid >> 5;             // wave owns cols [wave*128, +128)
  const int gsel = lane >> 4;
  const int nl   = lane & 15;
  const int rowBase = blockIdx.x * BM2;

  v8f acc[8];
  #pragma unroll
  for (int t = 0; t < 8; ++t)
    acc[t] = (v8f){0.f,0.f,0.f,0.f,0.f,0.f,0.f,0.f};

  for (int k0 = 0; k0 < H_DIM; k0 += KT) {
    // B: 1024 cols x 32 k = 4096 16B chunks -> 16 per thread
    #pragma unroll 4
    for (int j = 0; j < 16; ++j) {
      const int chunk = tid + 256 * j;
      const int col = chunk >> 2, ck = (chunk & 3) * 8;
      copy16(&WoutT[(size_t)col * H_DIM + k0 + ck], &sm.b[col][ck]);
    }
    // A: 16 rows x 32 k = 64 chunks
    if (tid < 64) {
      const int row = tid >> 2, ck = (tid & 3) * 8;
      copy16(&h_new_bf[(size_t)(rowBase + row) * H_DIM + k0 + ck], &Asm[row][ck]);
    }
    WAIT_ASYNC(0);
    __syncthreads();

    v16bf afrag;
    #pragma unroll
    for (int e = 0; e < 16; ++e) {
      const int k = 16 * gsel + e;
      afrag[e] = Asm[nl][k];
    }
    #pragma unroll
    for (int t = 0; t < 8; ++t) {
      v16bf bfrag;
      #pragma unroll
      for (int e = 0; e < 16; ++e)
        bfrag[e] = sm.b[wave * 128 + t * 16 + nl][16 * gsel + e];
      acc[t] = __builtin_amdgcn_wmma_f32_16x16x32_bf16(
          false, afrag, false, bfrag, (short)0, acc[t], false, false);
    }
    __syncthreads();
  }

  // park logits (+bias) in LDS (aliases B staging buffer — safe post-barrier)
  #pragma unroll
  for (int t = 0; t < 8; ++t) {
    const int n = wave * 128 + t * 16 + nl;
    const float b = bout[n];
    #pragma unroll
    for (int r = 0; r < 8; ++r)
      sm.logits[gsel * 8 + r][n] = acc[t][r] + b;
  }
  __syncthreads();

  // per-row log_softmax: 2 rows per wave, wave32 shuffle reductions
  #pragma unroll
  for (int rr = 0; rr < 2; ++rr) {
    const int m = wave * 2 + rr;
    float mx = -INFINITY;
    for (int nc = lane; nc < O_DIM; nc += 32)
      mx = fmaxf(mx, sm.logits[m][nc]);
    #pragma unroll
    for (int off = 16; off > 0; off >>= 1)
      mx = fmaxf(mx, __shfl_xor(mx, off, 32));
    float s = 0.f;
    for (int nc = lane; nc < O_DIM; nc += 32)
      s += __expf(sm.logits[m][nc] - mx);
    #pragma unroll
    for (int off = 16; off > 0; off >>= 1)
      s += __shfl_xor(s, off, 32);
    const float lse = mx + __logf(s);
    const size_t row = (size_t)(rowBase + m);
    for (int nc = lane; nc < O_DIM; nc += 32)
      out[row * O_DIM + nc] = sm.logits[m][nc] - lse;
  }
}

// ---------------------------------------------------------------------------
// Launch.  Workspace layout (bf16):
//   Acat  [16384][2048]  64 MB
//   WgT   [4096][2048]   16 MB
//   WoutT [1024][1024]    2 MB
//   HnewB [16384][1024]  32 MB          total ~114 MB
// ---------------------------------------------------------------------------
extern "C" void kernel_launch(void* const* d_in, const int* in_sizes, int n_in,
                              void* d_out, int out_size, void* d_ws, size_t ws_size,
                              hipStream_t stream) {
  (void)in_sizes; (void)n_in; (void)out_size; (void)ws_size;
  const float* x    = (const float*)d_in[0];
  const float* h    = (const float*)d_in[1];
  const float* c    = (const float*)d_in[2];
  const float* Wg   = (const float*)d_in[3];
  const float* bg   = (const float*)d_in[4];
  const float* Wout = (const float*)d_in[5];
  const float* bout = (const float*)d_in[6];

  float* out   = (float*)d_out;
  float* h_new = out + (size_t)B_DIM * O_DIM;
  float* c_new = h_new + (size_t)B_DIM * H_DIM;

  __bf16* Acat  = (__bf16*)d_ws;
  __bf16* WgT   = Acat  + (size_t)B_DIM * K_SUM;
  __bf16* WoutT = WgT   + (size_t)FOUR_H * K_SUM;
  __bf16* HnewB = WoutT + (size_t)O_DIM * H_DIM;

  // pre-passes: convert (and transpose weights) once
  cvt_cat_kernel<<<(B_DIM * (size_t)K_SUM) / (256 * 8), 256, 0, stream>>>(x, h, Acat);
  tconv_kernel<<<dim3(FOUR_H / 32, K_SUM / 32), 256, 0, stream>>>(Wg, WgT, K_SUM, FOUR_H);
  tconv_kernel<<<dim3(O_DIM / 32, H_DIM / 32), 256, 0, stream>>>(Wout, WoutT, H_DIM, O_DIM);

  dim3 grid1(B_DIM / BM1, H_DIM / BN1);   // 128 x 32
  lstm_gates_kernel<<<grid1, 256, 0, stream>>>(Acat, WgT, c, bg, h_new, c_new, HnewB);

  out_proj_kernel<<<B_DIM / BM2, 256, 0, stream>>>(HnewB, WoutT, bout, out);
}
